// BasicDeconvolutionBlock_90615220011115
// MI455X (gfx1250) — matrix operand order repaired and checked by reading, not compiled, not run on hardware
//
#include <hip/hip_runtime.h>
#include <hip/hip_bf16.h>
#include <stdint.h>

// ---------------- problem constants (match reference) ----------------
#define N_IN   200000
#define N_OUT  600000
#define KOFF   27
#define PP     150000
#define CIN    96
#define COUT   96
#define BN_EPS 1e-5f

// tiling
#define ROWS_PER_WG 128                         // 8 waves * 16 rows
#define NBLK ((PP + ROWS_PER_WG - 1) / ROWS_PER_WG)   // 1172
#define WGS_PER_K 96

typedef __attribute__((ext_vector_type(16))) __bf16 v16bf;
typedef __attribute__((ext_vector_type(8)))  float  v8f;

__device__ __forceinline__ void atomic_add_f32(float* p, float v) {
    (void)__hip_atomic_fetch_add(p, v, __ATOMIC_RELAXED, __HIP_MEMORY_SCOPE_AGENT);
}

// ---------------- kernel 1: zero accumulator + stats ----------------
__global__ void zero_kernel(float* __restrict__ out, float* __restrict__ ws) {
    size_t i = (size_t)blockIdx.x * blockDim.x + threadIdx.x;
    const size_t tot4 = (size_t)N_OUT * COUT / 4;   // 14,400,000 float4
    if (i < tot4) {
        ((float4*)out)[i] = make_float4(0.f, 0.f, 0.f, 0.f);
    }
    if (i < 2 * COUT) ws[i] = 0.f;                  // sum / sumsq slots
}

// ---------------- kernel 2: gather -> WMMA GEMM -> scatter-add ----------------
__global__ __launch_bounds__(256) void spconv_scatter(
    const float* __restrict__ feats,
    const int*   __restrict__ in_idx,
    const int*   __restrict__ out_idx,
    const float* __restrict__ weight,
    float*       __restrict__ out)
{
    // transposed weights, split precision: w = hi + lo (bf16 each)
    __shared__ __attribute__((aligned(32))) __bf16 wh[CIN * COUT];
    __shared__ __attribute__((aligned(32))) __bf16 wl[CIN * COUT];
    // per-wave staged A rows (gathered via async-to-LDS)
    __shared__ __attribute__((aligned(16))) float As[8][16 * CIN];

    const int k      = blockIdx.x / WGS_PER_K;
    const int wgslot = blockIdx.x % WGS_PER_K;
    const int tid    = threadIdx.x;
    const int lane   = tid & 31;
    const int wv     = tid >> 5;
    const int half   = lane >> 4;     // 0 or 1
    const int r      = lane & 15;     // row within M-tile / col within N-tile

    // ---- stage weight[k] transposed as bf16 hi/lo (once per WG) ----
    const float* wk = weight + (size_t)k * (CIN * COUT);
    for (int i = tid; i < CIN * COUT; i += 256) {
        int ci = i / COUT, co = i % COUT;           // weight[k][ci][co]
        float w  = wk[i];
        __bf16 h = (__bf16)w;
        float lo = w - (float)h;
        wh[co * CIN + ci] = h;
        wl[co * CIN + ci] = (__bf16)lo;
    }
    __syncthreads();

    const int* ink  = in_idx  + (size_t)k * PP;
    const int* outk = out_idx + (size_t)k * PP;

    for (int blk = wgslot; blk < NBLK; blk += WGS_PER_K) {
        const int rowbase = blk * ROWS_PER_WG + wv * 16;

        // Make sure all DS reads of As from the previous iteration have
        // completed before async-LDS writes can overwrite the slab (WAR):
        // DS and async-LDS ops are not mutually ordered in hardware.
        asm volatile("s_wait_dscnt 0x0" ::: "memory");

        // ---- async gather: 16 rows x 96 fp32 (6144B) into this wave's LDS slab ----
        // 384 x 16B chunks, 32 lanes -> 12 issues per lane; ASYNCcnt tracked.
        #pragma unroll
        for (int it = 0; it < 12; ++it) {
            int chunk = it * 32 + lane;       // 0..383
            int rr    = chunk / 24;           // row 0..15
            int piece = chunk % 24;           // 16B piece within row
            int pr    = rowbase + rr;
            int src   = (pr < PP) ? ink[pr] : 0;
            uint64_t ga = (uint64_t)(const void*)(feats + (size_t)src * CIN + piece * 4);
            uint32_t la = (uint32_t)(uintptr_t)(&As[wv][rr * CIN + piece * 4]);
            asm volatile("global_load_async_to_lds_b128 %0, %1, off"
                         :: "v"(la), "v"(ga) : "memory");
        }
        asm volatile("s_wait_asynccnt 0x0" ::: "memory");  // wave-private region: no barrier needed

        v8f acc[6];
        #pragma unroll
        for (int t = 0; t < 6; ++t) {
            acc[t] = (v8f){0.f, 0.f, 0.f, 0.f, 0.f, 0.f, 0.f, 0.f};
        }

        const float* ar = &As[wv][r * CIN];
        #pragma unroll
        for (int c = 0; c < 3; ++c) {         // K-chunks of 32 input channels
            // A fragment, contiguous K order (matches B):
            //   e in 0..15 -> K = c*32 + half*16 + e
            v16bf ah, al;
            #pragma unroll
            for (int e = 0; e < 8; ++e) {
                float x1 = ar[c * 32 + half * 16 + e];
                float x2 = ar[c * 32 + half * 16 + 8 + e];
                __bf16 h1 = (__bf16)x1;
                __bf16 h2 = (__bf16)x2;
                ah[e]     = h1;  al[e]     = (__bf16)(x1 - (float)h1);
                ah[8 + e] = h2;  al[8 + e] = (__bf16)(x2 - (float)h2);
            }
            #pragma unroll
            for (int t = 0; t < 6; ++t) {     // N-tiles of 16 output channels
                // B fragment: col = t*16 + r, K = c*32 + half*16 + (0..15): 32B contiguous
                int woff = (t * 16 + r) * CIN + c * 32 + half * 16;
                v16bf bh = *(const v16bf*)&wh[woff];
                v16bf bl = *(const v16bf*)&wl[woff];
                // split-precision product: AhBh + AhBl + AlBh (~fp32 accuracy, fp32 accum)
                acc[t] = __builtin_amdgcn_wmma_f32_16x16x32_bf16(
                             false, ah, false, bh, (short)0, acc[t], false, false);
                acc[t] = __builtin_amdgcn_wmma_f32_16x16x32_bf16(
                             false, ah, false, bl, (short)0, acc[t], false, false);
                acc[t] = __builtin_amdgcn_wmma_f32_16x16x32_bf16(
                             false, al, false, bh, (short)0, acc[t], false, false);
            }
        }

        // ---- scatter-add: C/D layout row m = half*8 + v, col = t*16 + r ----
        // Non-returning global_atomic_add_f32 (result unused): STOREcnt,
        // fire-and-forget; the L2 atomic units absorb the reduction.
        #pragma unroll
        for (int v = 0; v < 8; ++v) {
            int m  = half * 8 + v;
            int pr = rowbase + m;
            if (pr < PP) {
                size_t orow = (size_t)outk[pr] * COUT;
                #pragma unroll
                for (int t = 0; t < 6; ++t) {
                    atomic_add_f32(&out[orow + t * 16 + r], acc[t][v]);
                }
            }
        }
    }
}

// ---------------- kernel 3: per-channel sum / sumsq ----------------
__global__ void col_stats(const float* __restrict__ out, float* __restrict__ ws) {
    const int c = threadIdx.x;                 // blockDim.x == 96, coalesced over a row
    float s = 0.f, s2 = 0.f;
    for (int rr = blockIdx.x; rr < N_OUT; rr += gridDim.x) {
        float v = out[(size_t)rr * COUT + c];
        s  += v;
        s2 += v * v;
    }
    atomic_add_f32(&ws[c], s);
    atomic_add_f32(&ws[COUT + c], s2);
}

// ---------------- kernel 4: fold BN into per-channel scale/shift ----------------
__global__ void bn_finalize(float* __restrict__ ws,
                            const float* __restrict__ gamma,
                            const float* __restrict__ beta) {
    const int c = threadIdx.x;                 // 96 threads, 1 block
    const float inv_n = 1.0f / (float)N_OUT;
    float mean = ws[c] * inv_n;
    float var  = ws[COUT + c] * inv_n - mean * mean;
    float sc   = gamma[c] * rsqrtf(var + BN_EPS);
    ws[2 * COUT + c] = sc;
    ws[3 * COUT + c] = beta[c] - mean * sc;
}

// ---------------- kernel 5: apply BN + ReLU in place (float4) ----------------
__global__ void bn_apply(float* __restrict__ out, const float* __restrict__ ws) {
    const size_t tot4 = (size_t)N_OUT * COUT / 4;  // 24 float4 per row
    size_t i = (size_t)blockIdx.x * blockDim.x + threadIdx.x;
    for (; i < tot4; i += (size_t)gridDim.x * blockDim.x) {
        int cc = (int)(i % 24) * 4;                // channel base of this float4
        float4 x  = ((const float4*)out)[i];
        float4 sc = *(const float4*)&ws[2 * COUT + cc];
        float4 sh = *(const float4*)&ws[3 * COUT + cc];
        x.x = fmaxf(fmaf(x.x, sc.x, sh.x), 0.f);
        x.y = fmaxf(fmaf(x.y, sc.y, sh.y), 0.f);
        x.z = fmaxf(fmaf(x.z, sc.z, sh.z), 0.f);
        x.w = fmaxf(fmaf(x.w, sc.w, sh.w), 0.f);
        ((float4*)out)[i] = x;
    }
}

// ---------------- launcher ----------------
extern "C" void kernel_launch(void* const* d_in, const int* in_sizes, int n_in,
                              void* d_out, int out_size, void* d_ws, size_t ws_size,
                              hipStream_t stream) {
    const float* feats  = (const float*)d_in[0];
    const int*   in_idx = (const int*)  d_in[1];
    const int*   out_idx= (const int*)  d_in[2];
    const float* weight = (const float*)d_in[3];
    const float* gamma  = (const float*)d_in[4];
    const float* beta   = (const float*)d_in[5];
    float* out = (float*)d_out;
    float* ws  = (float*)d_ws;

    const int tot4 = N_OUT * COUT / 4;                 // 14,400,000
    const int zgrid = (tot4 + 255) / 256;              // 56,250

    zero_kernel   <<<zgrid, 256, 0, stream>>>(out, ws);
    spconv_scatter<<<KOFF * WGS_PER_K, 256, 0, stream>>>(feats, in_idx, out_idx, weight, out);
    col_stats     <<<2048, 96, 0, stream>>>(out, ws);
    bn_finalize   <<<1, 96, 0, stream>>>(ws, gamma, beta);
    bn_apply      <<<zgrid, 256, 0, stream>>>(out, ws);
}
